// KDAAttention_74079595921665
// MI455X (gfx1250) — matrix operand order repaired and checked by reading, not compiled, not run on hardware
//
#include <hip/hip_runtime.h>
#include <math.h>

// ---------------- problem constants (match reference) ----------------
#define D_MODEL 1024
#define N_HEADS 16
#define D_HEAD  64
#define BB      4
#define TTLEN   2048
#define HD      (N_HEADS * D_HEAD)   // 1024
#define MROWS   (BB * TTLEN)         // 8192

// ---------------- vector types (native ext vectors: trivially copyable) ----------------
typedef __attribute__((ext_vector_type(16))) __bf16        v16bf;
typedef __attribute__((ext_vector_type(8)))  float         v8f;
typedef __attribute__((ext_vector_type(4)))  unsigned int  u32x4;
typedef __attribute__((ext_vector_type(2)))  unsigned int  u32x2;
typedef __attribute__((ext_vector_type(4)))  float         f32x4;
typedef __attribute__((ext_vector_type(8)))  int           i32x8;
typedef __attribute__((ext_vector_type(4)))  int           i32x4;

__device__ __forceinline__ unsigned short f32_to_bf16(float f) {
    unsigned u = __float_as_uint(f);
    u += 0x7FFFu + ((u >> 16) & 1u);   // round-to-nearest-even
    return (unsigned short)(u >> 16);
}

union V16 { u32x4 q[2]; v16bf v; };    // 32 bytes both ways

// ---------------- elementwise f32 -> bf16 conversion (vector x4) ----------------
__global__ void cvt_f32_bf16_v4(const float* __restrict__ in,
                                unsigned short* __restrict__ out, int n4)
{
    int i = blockIdx.x * 256 + threadIdx.x;
    if (i >= n4) return;
    f32x4 f = reinterpret_cast<const f32x4*>(in)[i];
    u32x2 o;
    o.x = (unsigned)f32_to_bf16(f.x) | ((unsigned)f32_to_bf16(f.y) << 16);
    o.y = (unsigned)f32_to_bf16(f.z) | ((unsigned)f32_to_bf16(f.w) << 16);
    reinterpret_cast<u32x2*>(out)[i] = o;
}

// ---------------- bf16 WMMA GEMM: Y = A * B^T + bias, optional sigmoid ----------------
// A: M x K bf16 row-major, Bw: N x K bf16 row-major, Y: M x N f32.
// Block tile 64(M) x 128(N), K stepped by 32. 256 threads = 8 waves (2x4).
// A tile is staged by the Tensor Data Mover (one descriptor per K step,
// issued by wave 0; TDM pad feature reproduces the +16B LDS row padding).
// Requires M%64==0, N%128==0, K%32==0 (true for all calls here).
#define TILE_M 64
#define TILE_N 128
#define TILE_K 32
#define APAD   8      // ushorts: 64B data + 16B pad per row (TDM pad_interval=3, pad_amount=3)
#define BPAD   8

__global__ __launch_bounds__(256)
void gemm_bias_act_wmma(const unsigned short* __restrict__ Abf,
                        const unsigned short* __restrict__ Bbf,
                        const float* __restrict__ bias,
                        float* __restrict__ Y,
                        int M, int N, int K, int act)
{
    __shared__ __align__(16) unsigned short As [TILE_M][TILE_K + APAD]; // row-major A tile
    __shared__ __align__(16) unsigned short BsT[TILE_K][TILE_N + BPAD]; // B tile, K-major

    const int tid  = threadIdx.x;
    const int lane = tid & 31;
    const int wave = tid >> 5;     // 0..7
    const int wm   = wave >> 2;    // 0..1  -> 32-row slice
    const int wn   = wave & 3;     // 0..3  -> 32-col slice

    const int m0 = blockIdx.y * TILE_M;
    const int n0 = blockIdx.x * TILE_N;

    v8f acc[2][2];
    #pragma unroll
    for (int a = 0; a < 2; ++a)
        #pragma unroll
        for (int c = 0; c < 2; ++c)
            #pragma unroll
            for (int r = 0; r < 8; ++r) acc[a][c][r] = 0.0f;

    const int mrow = lane & 15;    // M within 16x16 tile
    const int kgrp = lane >> 4;    // 0/1: K-half select for A fragment

    // LDS base of As for the TDM descriptor (low 32 bits of the flat shared addr)
    const unsigned ldsA = (unsigned)(unsigned long long)(void*)&As[0][0];

    for (int k0 = 0; k0 < K; k0 += TILE_K) {
        // ---- stage A tile 64x32 via Tensor Data Mover (wave 0 issues) ----
        if (wave == 0) {
            const unsigned long long ga =
                (unsigned long long)(const void*)(Abf + (size_t)m0 * K + k0);
            u32x4 g0;
            g0.x = 1u;                                            // count=1, user desc
            g0.y = ldsA;                                          // lds_addr
            g0.z = (unsigned)(ga & 0xFFFFFFFFull);                // global_addr[31:0]
            g0.w = (unsigned)((ga >> 32) & 0x01FFFFFFull)         // global_addr[56:32]
                 | (2u << 30);                                    // type=2 ("image")
            i32x8 g1;
            g1[0] = (int)((1u << 16)        // data_size = 1 -> 2 bytes
                        | (1u << 20)        // pad_enable
                        | (3u << 22)        // pad_interval: 16 DWORDs (= 32 bf16 row)
                        | (3u << 25));      // pad_amount:   4 DWORDs (= 16B)
            g1[1] = (int)(((unsigned)K & 0xFFFFu) << 16);                    // tensor_dim0 lo
            g1[2] = (int)(((unsigned)K >> 16) | (((unsigned)M & 0xFFFFu) << 16)); // dim0 hi | dim1 lo
            g1[3] = (int)(((unsigned)M >> 16) | ((unsigned)TILE_K << 16));   // dim1 hi | tile_dim0
            g1[4] = TILE_M;                                                  // tile_dim1 (tile_dim2=0)
            g1[5] = K;                                                       // tensor_dim0_stride lo
            g1[6] = 0;
            g1[7] = 0;
            i32x4 g2 = {0, 0, 0, 0};
            i32x4 g3 = {0, 0, 0, 0};
            i32x8 g4 = {0, 0, 0, 0, 0, 0, 0, 0};
            __builtin_amdgcn_tensor_load_to_lds(g0, g1, g2, g3, g4, 0);
        }

        // ---- stage B tile 128x32 transposed into BsT[k][n] ----
        #pragma unroll
        for (int it = 0; it < 2; ++it) {
            int idx = it * 256 + tid;       // 512 chunks of 8
            int r = idx >> 2;               // n row 0..127
            int c = (idx & 3) * 8;          // k col 0,8,16,24
            u32x4 wv = *reinterpret_cast<const u32x4*>(Bbf + (size_t)(n0 + r) * K + k0 + c);
            const unsigned short* p = reinterpret_cast<const unsigned short*>(&wv);
            #pragma unroll
            for (int e = 0; e < 8; ++e) BsT[c + e][r] = p[e];
        }

        // prefetch next K tile (global_prefetch_b8)
        if (k0 + TILE_K < K) {
            __builtin_prefetch(Abf + (size_t)(m0 + (tid >> 2)) * K + k0 + TILE_K, 0, 1);
            __builtin_prefetch(Bbf + (size_t)(n0 + (tid >> 1)) * K + k0 + TILE_K, 0, 1);
        }

        __builtin_amdgcn_s_wait_tensorcnt(0);   // TDM done (no-op for waves 1..7)
        __syncthreads();

        #pragma unroll
        for (int mi = 0; mi < 2; ++mi) {
            // A fragment per ISA 16-bit A 16x32 layout:
            // lane%16 = M; lane/16 selects K-halves {kb..kb+7, kb+16..kb+23}, kb=8*(lane/16)
            const int row = wm * 32 + mi * 16 + mrow;
            const int kb  = kgrp * 8;
            V16 af;
            af.q[0] = *reinterpret_cast<const u32x4*>(&As[row][kb]);
            af.q[1] = *reinterpret_cast<const u32x4*>(&As[row][kb + 16]);

            #pragma unroll
            for (int ni = 0; ni < 2; ++ni) {
                // B fragment (32x16): lane%16 = N, elements = K {kb..kb+7, kb+16..kb+23} (same as A)
                const int cb = wn * 32 + ni * 16;
                const int ncol = cb + mrow;
                V16 bf;
                #pragma unroll
                for (int e = 0; e < 4; ++e) {
                    bf.q[0][e] = (unsigned)BsT[kb + 2 * e][ncol] | ((unsigned)BsT[kb + 2 * e + 1][ncol] << 16);
                    bf.q[1][e] = (unsigned)BsT[kb + 16 + 2 * e][ncol] | ((unsigned)BsT[kb + 17 + 2 * e][ncol] << 16);
                }

                acc[mi][ni] = __builtin_amdgcn_wmma_f32_16x16x32_bf16(
                    false, af.v, false, bf.v, (short)0, acc[mi][ni], false, false);
            }
        }
        __syncthreads();
    }

    // Epilogue: C/D layout -> VGPR r: M = r + 8*(lane>=16), N = lane%16
    const int nlane = lane & 15;
    const int rbase = kgrp * 8;
    #pragma unroll
    for (int mi = 0; mi < 2; ++mi) {
        #pragma unroll
        for (int ni = 0; ni < 2; ++ni) {
            #pragma unroll
            for (int r = 0; r < 8; ++r) {
                const int m = m0 + wm * 32 + mi * 16 + rbase + r;
                const int n = n0 + wn * 32 + ni * 16 + nlane;
                float v = acc[mi][ni][r] + bias[n];
                if (act == 1) v = __builtin_amdgcn_rcpf(1.0f + __expf(-v));  // sigmoid
                Y[(size_t)m * N + n] = v;
            }
        }
    }
}

// ---------------- beta projection: softplus(X * Wb^T + bb), N = 16 ----------------
__global__ void beta_proj_softplus(const float* __restrict__ X,
                                   const float* __restrict__ Wb,
                                   const float* __restrict__ bb,
                                   float* __restrict__ Beta,
                                   int M, int K)
{
    int g = blockIdx.x * blockDim.x + threadIdx.x;
    if (g >= M * N_HEADS) return;
    int m = g >> 4, h = g & 15;
    const f32x4* xr = reinterpret_cast<const f32x4*>(X  + (size_t)m * K);
    const f32x4* wr = reinterpret_cast<const f32x4*>(Wb + (size_t)h * K);
    float s = 0.0f;
    for (int k = 0; k < K / 4; ++k) {
        f32x4 a = xr[k], w = wr[k];
        s += a.x * w.x + a.y * w.y + a.z * w.z + a.w * w.w;
    }
    s += bb[h];
    Beta[g] = (s > 20.0f) ? s : log1pf(__expf(s));   // softplus
}

// ---------------- gated delta-rule scan ----------------
// One block per (b,h). 256 threads hold the 64x64 fp32 state in registers:
// thread t owns column j = t&63, rows i in [16*(t>>6), +16).
__global__ __launch_bounds__(256)
void kda_scan(const float* __restrict__ Q, const float* __restrict__ Kp,
              const float* __restrict__ V, const float* __restrict__ A,
              const float* __restrict__ Beta,
              unsigned short* __restrict__ Obf,    // bf16 output for final GEMM
              float* __restrict__ Sfinal)
{
    const int bh = blockIdx.x;      // 0..63
    const int b  = bh >> 4;
    const int h  = bh & 15;
    const int tid = threadIdx.x;
    const int j   = tid & 63;
    const int iq  = tid >> 6;       // 0..3

    __shared__ float kv[64], vv[64], qv[64], av[64];
    __shared__ float partS[4][64];
    __shared__ float partO[4][64];
    __shared__ float bsh;

    float S[16];
    #pragma unroll
    for (int r = 0; r < 16; ++r) S[r] = 0.0f;

    for (int t = 0; t < TTLEN; ++t) {
        __syncthreads();
        const size_t off = ((size_t)b * TTLEN + t) * (size_t)HD + h * D_HEAD + j;
        if      (iq == 0) kv[j] = Kp[off];
        else if (iq == 1) vv[j] = V[off];
        else if (iq == 2) qv[j] = Q[off];
        else              av[j] = A[off];
        if (tid == 0) bsh = Beta[((size_t)b * TTLEN + t) * N_HEADS + h];
        __syncthreads();

        const float bt = bsh;
        float p = 0.0f;
        #pragma unroll
        for (int r = 0; r < 16; ++r) {
            const int i = iq * 16 + r;
            S[r] *= av[i];              // decay
            p    += kv[i] * S[r];       // partial k^T S
        }
        partS[iq][j] = p;
        __syncthreads();

        const float kts  = partS[0][j] + partS[1][j] + partS[2][j] + partS[3][j];
        const float corr = bt * (vv[j] - kts);
        float po = 0.0f;
        #pragma unroll
        for (int r = 0; r < 16; ++r) {
            const int i = iq * 16 + r;
            S[r] += kv[i] * corr;       // rank-1 delta update
            po   += qv[i] * S[r];       // partial q^T S_new
        }
        partO[iq][j] = po;
        __syncthreads();

        if (iq == 0)
            Obf[((size_t)b * TTLEN + t) * (size_t)HD + h * D_HEAD + j] =
                f32_to_bf16(partO[0][j] + partO[1][j] + partO[2][j] + partO[3][j]);
    }

    // Final state (B,H,64,64) -> second tuple output (fp32)
    #pragma unroll
    for (int r = 0; r < 16; ++r)
        Sfinal[(((size_t)bh) * 64 + iq * 16 + r) * 64 + j] = S[r];
}

// ---------------- launcher ----------------
extern "C" void kernel_launch(void* const* d_in, const int* in_sizes, int n_in,
                              void* d_out, int out_size, void* d_ws, size_t ws_size,
                              hipStream_t stream)
{
    const float* x  = (const float*)d_in[0];
    const float* Wq = (const float*)d_in[1];
    const float* bq = (const float*)d_in[2];
    const float* Wk = (const float*)d_in[3];
    const float* bk = (const float*)d_in[4];
    const float* Wv = (const float*)d_in[5];
    const float* bv = (const float*)d_in[6];
    const float* Wa = (const float*)d_in[7];
    const float* ba = (const float*)d_in[8];
    const float* Wb = (const float*)d_in[9];
    const float* bb = (const float*)d_in[10];
    const float* Wo = (const float*)d_in[11];
    const float* bo = (const float*)d_in[12];

    float* y      = (float*)d_out;                          // (B,T,D_MODEL)
    float* Sfinal = y + (size_t)MROWS * D_MODEL;            // (B,H,64,64)

    // ---- workspace layout ----
    float* ws = (float*)d_ws;
    const size_t SZ = (size_t)MROWS * HD;                   // 8192*1024
    float* Qp = ws;                                         // f32 activations
    float* Kp = Qp + SZ;
    float* Vp = Kp + SZ;
    float* Ap = Vp + SZ;
    float* Bp = Ap + SZ;                                    // beta (MROWS*16) f32
    unsigned short* xbf  = (unsigned short*)(Bp + (size_t)MROWS * N_HEADS);
    unsigned short* Wqbf = xbf  + (size_t)MROWS * D_MODEL;
    unsigned short* Wkbf = Wqbf + (size_t)HD * D_MODEL;
    unsigned short* Wvbf = Wkbf + (size_t)HD * D_MODEL;
    unsigned short* Wabf = Wvbf + (size_t)HD * D_MODEL;
    unsigned short* Wobf = Wabf + (size_t)HD * D_MODEL;
    unsigned short* Obf  = Wobf + (size_t)D_MODEL * HD;

    dim3 blk(256);

    // ---- one-time f32 -> bf16 conversions ----
    {
        int nx4 = (MROWS * D_MODEL) / 4;
        int nw4 = (HD * D_MODEL) / 4;
        cvt_f32_bf16_v4<<<(nx4 + 255) / 256, blk, 0, stream>>>(x,  xbf,  nx4);
        cvt_f32_bf16_v4<<<(nw4 + 255) / 256, blk, 0, stream>>>(Wq, Wqbf, nw4);
        cvt_f32_bf16_v4<<<(nw4 + 255) / 256, blk, 0, stream>>>(Wk, Wkbf, nw4);
        cvt_f32_bf16_v4<<<(nw4 + 255) / 256, blk, 0, stream>>>(Wv, Wvbf, nw4);
        cvt_f32_bf16_v4<<<(nw4 + 255) / 256, blk, 0, stream>>>(Wa, Wabf, nw4);
        cvt_f32_bf16_v4<<<(nw4 + 255) / 256, blk, 0, stream>>>(Wo, Wobf, nw4);
    }

    // ---- projections (bf16 WMMA, f32 accumulate, TDM-staged A tiles) ----
    dim3 g_proj(HD / TILE_N, MROWS / TILE_M);               // (8, 128)
    gemm_bias_act_wmma<<<g_proj, blk, 0, stream>>>(xbf, Wqbf, bq, Qp, MROWS, HD, D_MODEL, 0);
    gemm_bias_act_wmma<<<g_proj, blk, 0, stream>>>(xbf, Wkbf, bk, Kp, MROWS, HD, D_MODEL, 0);
    gemm_bias_act_wmma<<<g_proj, blk, 0, stream>>>(xbf, Wvbf, bv, Vp, MROWS, HD, D_MODEL, 0);
    gemm_bias_act_wmma<<<g_proj, blk, 0, stream>>>(xbf, Wabf, ba, Ap, MROWS, HD, D_MODEL, 1);

    beta_proj_softplus<<<(MROWS * N_HEADS + 255) / 256, blk, 0, stream>>>(x, Wb, bb, Bp,
                                                                          MROWS, D_MODEL);

    // ---- recurrent scan (state resident in registers) ----
    kda_scan<<<BB * N_HEADS, blk, 0, stream>>>(Qp, Kp, Vp, Ap, Bp, Obf, Sfinal);

    // ---- output projection ----
    dim3 g_out(D_MODEL / TILE_N, MROWS / TILE_M);
    gemm_bias_act_wmma<<<g_out, blk, 0, stream>>>(Obf, Wobf, bo, y, MROWS, D_MODEL, HD, 0);
}
